// AdaptiveGaussianWindowAttention_64544768524823
// MI455X (gfx1250) — matrix operand order repaired and checked by reading, not compiled, not run on hardware
//
#include <hip/hip_runtime.h>
#include <hip/hip_bf16.h>
#include <math.h>

#define DMODEL 1024
#define NHEADS 16
#define DHEAD  64
#define BATCH  2
#define SEQ    2048
#define MROWS  (BATCH * SEQ)   // 4096

typedef __attribute__((ext_vector_type(16))) __bf16 v16bf;
typedef __attribute__((ext_vector_type(8)))  float  v8f;

// ---- WMMA fragment loaders (wave32 layouts per CDNA5 ISA 7.12.2) ----------
// A 16x32 bf16: lane = 16*half + m ; element e -> K = e + 8*half + (e>=8 ? 8 : 0)
__device__ __forceinline__ v16bf load_a_bf16(const __bf16* X, int ld, int m0, int k0, int lane) {
  const int m = m0 + (lane & 15);
  const int h = lane >> 4;
  const __bf16* p0 = X + (size_t)m * ld + k0 + 8 * h;
  v16bf a;
#pragma unroll
  for (int i = 0; i < 8; ++i) {
    a[i]     = p0[i];
    a[i + 8] = p0[i + 16];
  }
  return a;
}

// A from fp32 with non-temporal (last-use) loads + inline cvt (attn stream only).
__device__ __forceinline__ v16bf load_a_f32_nt(const float* X, int ld, int m0, int k0, int lane) {
  const int m = m0 + (lane & 15);
  const int h = lane >> 4;
  const float* p0 = X + (size_t)m * ld + k0 + 8 * h;
  v16bf a;
#pragma unroll
  for (int i = 0; i < 8; ++i) {
    a[i]     = (__bf16)__builtin_nontemporal_load(&p0[i]);
    a[i + 8] = (__bf16)__builtin_nontemporal_load(&p0[i + 16]);
  }
  return a;
}

// B 32x16 from row-major bf16 [K x N]: lane = 16*half + n ; element e -> K = e + 8*half + (e>=8 ? 8 : 0)
__device__ __forceinline__ v16bf load_b_rm_bf16(const __bf16* Bm, int ld, int k0, int n0, int lane) {
  const int h = lane >> 4;
  const __bf16* p0 = Bm + (size_t)(k0 + 8 * h) * ld + n0 + (lane & 15);
  v16bf b;
#pragma unroll
  for (int i = 0; i < 8; ++i) {
    b[i]     = p0[(size_t)i * ld];
    b[i + 8] = p0[(size_t)(i + 16) * ld];
  }
  return b;
}

// ---- Kernel 0a: plain fp32 -> bf16 conversion (activations) ---------------
__global__ void cvt_bf16_kernel(const float* __restrict__ src, __bf16* __restrict__ dst) {
  const size_t i = ((size_t)blockIdx.x * blockDim.x + threadIdx.x) * 4;
  const float4 f = *(const float4*)(src + i);
  dst[i + 0] = (__bf16)f.x;
  dst[i + 1] = (__bf16)f.y;
  dst[i + 2] = (__bf16)f.z;
  dst[i + 3] = (__bf16)f.w;
}

// ---- Kernel 0b: fp32 [N x K] -> bf16 transposed [K x N] (weights) ---------
// 32x32 LDS tile transpose, coalesced on both sides.
__global__ void cvt_t_bf16_kernel(const float* __restrict__ src, __bf16* __restrict__ dst) {
  __shared__ float tile[32][33];
  const int n0 = blockIdx.x * 32;          // src row block
  const int k0 = blockIdx.y * 32;          // src col block
  const int tx = threadIdx.x;              // 0..31
  const int ty = threadIdx.y;              // 0..7
#pragma unroll
  for (int i = 0; i < 4; ++i) {
    const int r = ty * 4 + i;
    tile[r][tx] = src[(size_t)(n0 + r) * DMODEL + k0 + tx];
  }
  __syncthreads();
#pragma unroll
  for (int i = 0; i < 4; ++i) {
    const int r = ty * 4 + i;
    dst[(size_t)(k0 + r) * DMODEL + n0 + tx] = (__bf16)tile[tx][r];
  }
}

// ---- Kernel 1: projection  Y = x @ W^T + b (W given pre-transposed bf16) --
// MODE 0: store [B,H,L,Dh] (Q, V).  MODE 1: store [B,H,Dh,L] (K, transposed
// so the scores kernel gets contiguous K^T B-fragments).
template <int MODE>
__global__ void proj_kernel(const __bf16* __restrict__ X, const __bf16* __restrict__ Wt,
                            const float* __restrict__ bias, __bf16* __restrict__ Y) {
  const int lane = threadIdx.x & 31;
  const int wave = threadIdx.x >> 5;
  const int mbase = blockIdx.x * 64;
  const int n0 = blockIdx.y * 64 + wave * 16;
  v8f acc[4] = {};
#pragma unroll 2
  for (int k0 = 0; k0 < DMODEL; k0 += 32) {
    const v16bf b = load_b_rm_bf16(Wt, DMODEL, k0, n0, lane);
#pragma unroll
    for (int t = 0; t < 4; ++t) {
      v16bf a = load_a_bf16(X, DMODEL, mbase + 16 * t, k0, lane);
      acc[t] = __builtin_amdgcn_wmma_f32_16x16x32_bf16(false, a, false, b, (short)0, acc[t], false, false);
    }
  }
  const int half = lane >> 4;
  const int n  = n0 + (lane & 15);
  const int hd = n >> 6;           // head index
  const int d  = n & (DHEAD - 1);
  const float bn = bias[n];
#pragma unroll
  for (int t = 0; t < 4; ++t) {
#pragma unroll
    for (int r = 0; r < 8; ++r) {
      const int m  = mbase + 16 * t + r + 8 * half;
      const int b_ = m >> 11;            // batch (L = 2048)
      const int l  = m & (SEQ - 1);
      const __bf16 val = (__bf16)(acc[t][r] + bn);
      if (MODE == 0)
        Y[((size_t)(b_ * NHEADS + hd) * SEQ + l) * DHEAD + d] = val;
      else
        Y[((size_t)(b_ * NHEADS + hd) * DHEAD + d) * SEQ + l] = val;
    }
  }
}

// ---- Kernel 2: scores = QK^T/8 - lambda*(l-m)^2  -> fp32 attn buffer ------
// Kt is [B,H,Dh,L]: B-frags are contiguous. One Q A-frag feeds 4 WMMAs.
__global__ void scores_kernel(const __bf16* __restrict__ Qb, const __bf16* __restrict__ Kt,
                              const float* __restrict__ log_lambda, float* __restrict__ attn) {
  const int lane = threadIdx.x & 31;
  const int wave = threadIdx.x >> 5;
  const int bh = blockIdx.z;
  const int l0 = blockIdx.x * 16;
  const int mbase = blockIdx.y * 256 + wave * 64;
  const __bf16* Qh  = Qb + (size_t)bh * SEQ * DHEAD;
  const __bf16* Kth = Kt + (size_t)bh * SEQ * DHEAD;   // [Dh x L]
  v8f acc[4] = {};
#pragma unroll
  for (int k0 = 0; k0 < DHEAD; k0 += 32) {
    const v16bf a = load_a_bf16(Qh, DHEAD, l0, k0, lane);
#pragma unroll
    for (int u = 0; u < 4; ++u) {
      v16bf b = load_b_rm_bf16(Kth, SEQ, k0, mbase + 16 * u, lane);
      acc[u] = __builtin_amdgcn_wmma_f32_16x16x32_bf16(false, a, false, b, (short)0, acc[u], false, false);
    }
  }
  const float lam = __expf(log_lambda[bh & (NHEADS - 1)]);
  const int half = lane >> 4;
  float* S = attn + (size_t)bh * SEQ * SEQ;
#pragma unroll
  for (int u = 0; u < 4; ++u) {
    const int m = mbase + 16 * u + (lane & 15);
#pragma unroll
    for (int r = 0; r < 8; ++r) {
      const int l = l0 + r + 8 * half;
      const float diff = (float)(l - m);
      S[(size_t)l * SEQ + m] = acc[u][r] * 0.125f - lam * diff * diff;
    }
  }
}

// ---- Kernel 3: in-place row softmax over 2048 columns ---------------------
__global__ void softmax_kernel(float* __restrict__ attn) {
  __shared__ float red[256];
  float* row = attn + (size_t)blockIdx.x * SEQ;
  const int t = threadIdx.x;
  float v[8];
  float mx = -INFINITY;
#pragma unroll
  for (int i = 0; i < 8; ++i) { v[i] = row[t + i * 256]; mx = fmaxf(mx, v[i]); }
  red[t] = mx; __syncthreads();
  for (int s = 128; s > 0; s >>= 1) { if (t < s) red[t] = fmaxf(red[t], red[t + s]); __syncthreads(); }
  mx = red[0]; __syncthreads();
  float sum = 0.f;
#pragma unroll
  for (int i = 0; i < 8; ++i) { v[i] = __expf(v[i] - mx); sum += v[i]; }
  red[t] = sum; __syncthreads();
  for (int s = 128; s > 0; s >>= 1) { if (t < s) red[t] += red[t + s]; __syncthreads(); }
  const float inv = 1.f / red[0];
#pragma unroll
  for (int i = 0; i < 8; ++i) row[t + i * 256] = v[i] * inv;
}

// ---- Kernel 4: O = attn @ V  -> bf16, layout [B, L, H*Dh] -----------------
// One contiguous V B-frag feeds 4 WMMAs; attn reads NT (last use).
__global__ void av_kernel(const float* __restrict__ attn, const __bf16* __restrict__ Vb,
                          __bf16* __restrict__ Ob) {
  const int lane = threadIdx.x & 31;
  const int wave = threadIdx.x >> 5;
  const int bh = blockIdx.y;
  const int lbase = blockIdx.x * 64;
  const int n0 = wave * 16;                 // 4 waves cover Dh = 64
  const float* Ah = attn + (size_t)bh * SEQ * SEQ;
  const __bf16* Vh = Vb + (size_t)bh * SEQ * DHEAD;
  v8f acc[4] = {};
  for (int k0 = 0; k0 < SEQ; k0 += 32) {
    const v16bf b = load_b_rm_bf16(Vh, DHEAD, k0, n0, lane);
#pragma unroll
    for (int t = 0; t < 4; ++t) {
      v16bf a = load_a_f32_nt(Ah, SEQ, lbase + 16 * t, k0, lane);
      acc[t] = __builtin_amdgcn_wmma_f32_16x16x32_bf16(false, a, false, b, (short)0, acc[t], false, false);
    }
  }
  const int half = lane >> 4;
  const int b_ = bh >> 4, hd = bh & (NHEADS - 1);
  const int col = hd * DHEAD + n0 + (lane & 15);
#pragma unroll
  for (int t = 0; t < 4; ++t) {
#pragma unroll
    for (int r = 0; r < 8; ++r) {
      const int l = lbase + 16 * t + r + 8 * half;
      Ob[((size_t)(b_ * SEQ + l)) * DMODEL + col] = (__bf16)acc[t][r];
    }
  }
}

// ---- Kernel 5: out = O @ Wo^T + bo  -> fp32 [B, L, D] (Wo pre-transposed) -
__global__ void oproj_kernel(const __bf16* __restrict__ X, const __bf16* __restrict__ Wt,
                             const float* __restrict__ bias, float* __restrict__ Y) {
  const int lane = threadIdx.x & 31;
  const int wave = threadIdx.x >> 5;
  const int mbase = blockIdx.x * 64;
  const int n0 = blockIdx.y * 64 + wave * 16;
  v8f acc[4] = {};
#pragma unroll 2
  for (int k0 = 0; k0 < DMODEL; k0 += 32) {
    const v16bf b = load_b_rm_bf16(Wt, DMODEL, k0, n0, lane);
#pragma unroll
    for (int t = 0; t < 4; ++t) {
      v16bf a = load_a_bf16(X, DMODEL, mbase + 16 * t, k0, lane);
      acc[t] = __builtin_amdgcn_wmma_f32_16x16x32_bf16(false, a, false, b, (short)0, acc[t], false, false);
    }
  }
  const int half = lane >> 4;
  const int n = n0 + (lane & 15);
  const float bn = bias[n];
#pragma unroll
  for (int t = 0; t < 4; ++t) {
#pragma unroll
    for (int r = 0; r < 8; ++r) {
      const int m = mbase + 16 * t + r + 8 * half;
      Y[(size_t)m * DMODEL + n] = acc[t][r] + bn;
    }
  }
}

extern "C" void kernel_launch(void* const* d_in, const int* in_sizes, int n_in,
                              void* d_out, int out_size, void* d_ws, size_t ws_size,
                              hipStream_t stream) {
  const float* q  = (const float*)d_in[0];
  const float* k  = (const float*)d_in[1];
  const float* v  = (const float*)d_in[2];
  const float* Wq = (const float*)d_in[3];
  const float* bq = (const float*)d_in[4];
  const float* Wk = (const float*)d_in[5];
  const float* bk = (const float*)d_in[6];
  const float* Wv = (const float*)d_in[7];
  const float* bv = (const float*)d_in[8];
  const float* Wo = (const float*)d_in[9];
  const float* bo = (const float*)d_in[10];
  const float* log_lambda = (const float*)d_in[11];

  float* out  = (float*)d_out;                       // [B, L, D]
  float* attn = out + (size_t)BATCH * SEQ * DMODEL;  // [B, H, L, L] (also scores scratch)

  // workspace layout (48 MB total):
  const size_t ACT = (size_t)MROWS * DMODEL;         // 4M elements
  const size_t WEL = (size_t)DMODEL * DMODEL;        // 1M elements
  __bf16* qb   = (__bf16*)d_ws;                      // [B,H,L,Dh]  8MB
  __bf16* kt   = qb + ACT;                           // [B,H,Dh,L]  8MB
  __bf16* vb   = kt + ACT;                           // [B,H,L,Dh]  8MB
  __bf16* ob   = vb + ACT;                           // [B,L,D]     8MB
  __bf16* xbuf = ob + ACT;                           // staging     8MB
  __bf16* wqt  = xbuf + ACT;                         // W^T bf16, 4 x 2MB
  __bf16* wkt  = wqt + WEL;
  __bf16* wvt  = wkt + WEL;
  __bf16* wot  = wvt + WEL;

  // one-shot weight transpose+convert fp32 [NxK] -> bf16 [KxN]
  const dim3 tblk(32, 8);
  const dim3 tgrd(DMODEL / 32, DMODEL / 32);
  cvt_t_bf16_kernel<<<tgrd, tblk, 0, stream>>>(Wq, wqt);
  cvt_t_bf16_kernel<<<tgrd, tblk, 0, stream>>>(Wk, wkt);
  cvt_t_bf16_kernel<<<tgrd, tblk, 0, stream>>>(Wv, wvt);
  cvt_t_bf16_kernel<<<tgrd, tblk, 0, stream>>>(Wo, wot);

  const dim3 cblk(256);
  const dim3 gcx(ACT / 1024);                        // 4M / (256*4)
  const dim3 blk(128);                               // 4 waves (wave32)
  const dim3 gproj(MROWS / 64, DMODEL / 64);
  // stage each activation through xbuf (bf16), then all-bf16 projection GEMM
  cvt_bf16_kernel<<<gcx, cblk, 0, stream>>>(q, xbuf);
  proj_kernel<0><<<gproj, blk, 0, stream>>>(xbuf, wqt, bq, qb);
  cvt_bf16_kernel<<<gcx, cblk, 0, stream>>>(k, xbuf);
  proj_kernel<1><<<gproj, blk, 0, stream>>>(xbuf, wkt, bk, kt);   // K stored transposed
  cvt_bf16_kernel<<<gcx, cblk, 0, stream>>>(v, xbuf);
  proj_kernel<0><<<gproj, blk, 0, stream>>>(xbuf, wvt, bv, vb);

  const dim3 gsc(SEQ / 16, SEQ / 256, BATCH * NHEADS);
  scores_kernel<<<gsc, blk, 0, stream>>>(qb, kt, log_lambda, attn);

  softmax_kernel<<<dim3(BATCH * NHEADS * SEQ), dim3(256), 0, stream>>>(attn);

  const dim3 gav(SEQ / 64, BATCH * NHEADS);
  av_kernel<<<gav, blk, 0, stream>>>(attn, vb, ob);

  oproj_kernel<<<gproj, blk, 0, stream>>>(ob, wot, bo, out);
}
